// NSAAttention_17549236371863
// MI455X (gfx1250) — matrix operand order repaired and checked by reading, not compiled, not run on hardware
//
#include <hip/hip_runtime.h>

// ---------------- problem constants ----------------
#define N_    1024
#define DIM_  512
#define H_    8
#define KVH_  2
#define DH_   64
#define REP_  4
#define BLK_  64
#define STRIDE_ 32
#define NSEL_ 8
#define WIN_  64
#define NSB_  16
#define NCB_  31
#define NCTX_ 32           // NCB + 1 (mem slot)
#define QKVW_ 768          // (H + 2*KVH) * DH
#define NEGF  (-1e30f)
#define SCALE_ 0.125f      // DH^-0.5

typedef __bf16 bf16;
typedef __attribute__((ext_vector_type(16))) __bf16 v16bf;
typedef __attribute__((ext_vector_type(8)))  __bf16 v8bf;
typedef __attribute__((ext_vector_type(8)))  float  v8f;

__device__ __forceinline__ bf16 f2bf(float x) { return (bf16)x; }

// D = A(16x32 bf16) * B(32x16 bf16) + C(16x16 f32), wave32
__device__ __forceinline__ v8f wmma_bf16(v16bf a, v16bf b, v8f c) {
  return __builtin_amdgcn_wmma_f32_16x16x32_bf16(false, a, false, b, (short)0, c,
                                                 false, false);
}

// A fragment, row-major MxK tile (ld = row stride). Each lane loads one
// contiguous 16-element run: K {0-15} (lanes<16) or K {16-31} (lanes>=16),
// matching the B^T fragment order  ->  two b128 loads + shuffle.
__device__ __forceinline__ v16bf load_A(const bf16* base, int ld) {
  int lane = threadIdx.x & 31;
  const bf16* r = base + (lane & 15) * ld + ((lane >> 4) << 4);
  v8bf lo = *(const v8bf*)r;
  v8bf hi = *(const v8bf*)(r + 8);
  return __builtin_shufflevector(lo, hi, 0, 1, 2, 3, 4, 5, 6, 7,
                                 8, 9, 10, 11, 12, 13, 14, 15);
}

// B^T fragment: source is N-major (row-major NxK; rows = output column /
// key token / value dim). Per lane: one contiguous 16-element (32B) run.
__device__ __forceinline__ v16bf load_Bt(const bf16* base, int ld) {
  int lane = threadIdx.x & 31;
  return *(const v16bf*)(base + (lane & 15) * ld + ((lane >> 4) << 4));
}

// ---------------- small utility kernels ----------------
__global__ void k_cvt_t(const float* __restrict__ s, bf16* __restrict__ dT,
                        int K, int Nn) {  // s: K x Nn row-major -> dT: Nn x K
  int i = blockIdx.x * blockDim.x + threadIdx.x;
  if (i >= K * Nn) return;
  int k = i / Nn, n = i % Nn;
  dT[n * K + k] = f2bf(s[i]);
}

__global__ void k_zero(float* __restrict__ p, int n) {
  int i = blockIdx.x * blockDim.x + threadIdx.x;
  if (i < n) p[i] = 0.f;
}

// ---------------- RMSNorm ----------------
__global__ void k_rms(const float* __restrict__ x, const float* __restrict__ g,
                      float* __restrict__ xn, bf16* __restrict__ xnb) {
  int n = blockIdx.x;
  __shared__ float red[8];
  const float* row = x + n * DIM_;
  float s = 0.f;
  for (int i = threadIdx.x; i < DIM_; i += blockDim.x) { float v = row[i]; s += v * v; }
  for (int o = 16; o > 0; o >>= 1) s += __shfl_down(s, o, 32);
  if ((threadIdx.x & 31) == 0) red[threadIdx.x >> 5] = s;
  __syncthreads();
  if (threadIdx.x == 0) {
    float t = 0.f;
    for (int w2 = 0; w2 < (int)(blockDim.x >> 5); ++w2) t += red[w2];
    red[0] = rsqrtf(t / (float)DIM_ + 1e-6f);
  }
  __syncthreads();
  float r = red[0];
  for (int i = threadIdx.x; i < DIM_; i += blockDim.x) {
    float v = row[i] * r * g[i];
    xn[n * DIM_ + i] = v;
    xnb[n * DIM_ + i] = f2bf(v);
  }
}

// ---------------- generic bf16 WMMA GEMM: C(MxN) = A(MxK) @ Bt(NxK)^T + bias ----------------
__global__ void k_gemm(const bf16* __restrict__ A, const bf16* __restrict__ Bt,
                       const float* __restrict__ bias, float* __restrict__ C,
                       int M, int Nn, int K) {
  int wave = blockIdx.x * (blockDim.x >> 5) + (threadIdx.x >> 5);
  int ntiles = Nn >> 4;
  int mt = wave / ntiles, nt = wave % ntiles;
  if (mt * 16 >= M) return;
  int lane = threadIdx.x & 31;
  int arow = mt * 16 + (lane & 15);
  if (arow >= M) arow = M - 1;  // clamp; stores are guarded
  const bf16* arp = A + arow * K + ((lane >> 4) << 4);
  const bf16* brp = Bt + (nt * 16 + (lane & 15)) * K + ((lane >> 4) << 4);
  v8f acc = {};
  for (int kk = 0; kk < K; kk += 32) {
    v8bf a_lo = *(const v8bf*)(arp + kk);
    v8bf a_hi = *(const v8bf*)(arp + kk + 8);
    v16bf a = __builtin_shufflevector(a_lo, a_hi, 0, 1, 2, 3, 4, 5, 6, 7,
                                      8, 9, 10, 11, 12, 13, 14, 15);
    v16bf b = *(const v16bf*)(brp + kk);
    if (kk + 32 < K) __builtin_prefetch(brp + kk + 32, 0, 1);  // global_prefetch_b8
    acc = wmma_bf16(a, b, acc);
  }
  int nc = nt * 16 + (lane & 15);
  int mb = (lane >> 4) * 8;
  float bi = bias ? bias[nc] : 0.f;
#pragma unroll
  for (int r = 0; r < 8; ++r) {
    int m = mt * 16 + mb + r;
    if (m < M) C[m * Nn + nc] = acc[r] + bi;
  }
}

// ---------------- RoPE + head split; V stored dim-major for PV B-fragments ----------------
__global__ void k_rope(const float* __restrict__ qkv, bf16* __restrict__ rq,
                       bf16* __restrict__ qb, bf16* __restrict__ rk,
                       bf16* __restrict__ vT) {
  int head = blockIdx.x % (H_ + KVH_);
  int n = blockIdx.x / (H_ + KVH_);
  int i = threadIdx.x;  // 0..31 (pair i, i+32)
  float freq = __expf(-(float)i * (9.210340371976184f / 32.0f));  // 10000^(-i/32)
  float ang = (float)n * freq;
  float sn, cs;
  __sincosf(ang, &sn, &cs);
  if (head < H_) {
    const float* src = qkv + n * QKVW_ + head * DH_;
    float x1 = src[i], x2 = src[i + 32];
    bf16* o = rq + (head * N_ + n) * DH_;
    o[i] = f2bf(x1 * cs - x2 * sn);
    o[i + 32] = f2bf(x1 * sn + x2 * cs);
    bf16* oq = qb + (head * N_ + n) * DH_;
    oq[i] = f2bf(x1);
    oq[i + 32] = f2bf(x2);
  } else {
    int g = head - H_;
    const float* src = qkv + n * QKVW_ + H_ * DH_ + g * DH_;
    float x1 = src[i], x2 = src[i + 32];
    bf16* o = rk + (g * N_ + n) * DH_;
    o[i] = f2bf(x1 * cs - x2 * sn);
    o[i + 32] = f2bf(x1 * sn + x2 * cs);
    const float* vs = qkv + n * QKVW_ + (H_ + KVH_) * DH_ + g * DH_;
    vT[(g * DH_ + i) * N_ + n] = f2bf(vs[i]);
    vT[(g * DH_ + i + 32) * N_ + n] = f2bf(vs[i + 32]);
  }
}

// ---------------- build compressed-projection inputs ----------------
__global__ void k_build_cin(const float* __restrict__ qkv, const float* __restrict__ k_pos,
                            const float* __restrict__ v_pos, bf16* __restrict__ ckin,
                            bf16* __restrict__ cvin) {
  int row = blockIdx.x;  // g*NCB + cb
  int g = row / NCB_, cb = row % NCB_;
  int start = cb * STRIDE_;
  for (int j = threadIdx.x; j < BLK_ * DH_; j += blockDim.x) {
    int bp = j >> 6, d = j & 63;
    int nsrc = start + bp;
    float kv = qkv[nsrc * QKVW_ + H_ * DH_ + g * DH_ + d] + k_pos[(g * BLK_ + bp) * DH_ + d];
    float vv = qkv[nsrc * QKVW_ + (H_ + KVH_) * DH_ + g * DH_ + d] + v_pos[(g * BLK_ + bp) * DH_ + d];
    ckin[row * (BLK_ * DH_) + j] = f2bf(kv);
    cvin[row * (BLK_ * DH_) + j] = f2bf(vv);
  }
}

// ---------------- assemble compressed K (token-major) and V (dim-major) ----------------
__global__ void k_assemble(const float* __restrict__ tmpk, const float* __restrict__ tmpv,
                           const float* __restrict__ mem_ck, const float* __restrict__ mem_cv,
                           bf16* __restrict__ ckb, bf16* __restrict__ cvbT) {
  int i = blockIdx.x * blockDim.x + threadIdx.x;
  if (i >= KVH_ * NCTX_ * DH_) return;
  int d = i & 63, m = (i >> 6) & 31, g = i >> 11;
  float kv, vv;
  if (m == 0) { kv = mem_ck[g * DH_ + d]; vv = mem_cv[g * DH_ + d]; }
  else {
    kv = tmpk[(g * NCB_ + m - 1) * DH_ + d];
    vv = tmpv[(g * NCB_ + m - 1) * DH_ + d];
  }
  ckb[(g * NCTX_ + m) * DH_ + d] = f2bf(kv);
  cvbT[(g * DH_ + d) * NCTX_ + m] = f2bf(vv);
}

// ---------------- compressed attention: per (g,rep,16-query tile) wave ----------------
__global__ void __launch_bounds__(32) k_cattn(const bf16* __restrict__ qb,
                                              const bf16* __restrict__ ckb,
                                              const bf16* __restrict__ cvbT,
                                              float* __restrict__ c_out,
                                              float* __restrict__ imp_c) {
  int bid = blockIdx.x;
  int ntile = bid & 63, rep = (bid >> 6) & 3, g = bid >> 8;
  int h = g * REP_ + rep;
  int n0 = ntile * 16;
  int lane = threadIdx.x;
  __shared__ __attribute__((aligned(32))) float S[16][NCTX_];
  __shared__ __attribute__((aligned(32))) bf16 P[16][NCTX_];

  // scores: Q(16x64) @ Ck^T(64x32)
  for (int nt = 0; nt < 2; ++nt) {
    v8f acc = {};
#pragma unroll
    for (int kk = 0; kk < DH_; kk += 32) {
      v16bf a = load_A(qb + (h * N_ + n0) * DH_ + kk, DH_);
      v16bf b = load_Bt(ckb + (g * NCTX_ + nt * 16) * DH_ + kk, DH_);
      acc = wmma_bf16(a, b, acc);
    }
    int mcol = nt * 16 + (lane & 15);
    int mb = (lane >> 4) * 8;
    int cend = (mcol == 0) ? -1 : (mcol - 1) * STRIDE_ + BLK_ - 1;
#pragma unroll
    for (int r = 0; r < 8; ++r) {
      int qi = mb + r;
      S[qi][mcol] = ((n0 + qi) >= cend) ? acc[r] * SCALE_ : NEGF;
    }
  }
  __syncthreads();

  if (lane < 16) {
    int ng = n0 + lane;
    float mx = -3.4e38f;
    for (int m2 = 0; m2 < NCTX_; ++m2) mx = fmaxf(mx, S[lane][m2]);
    float sum = 0.f;
    for (int m2 = 0; m2 < NCTX_; ++m2) sum += __expf(S[lane][m2] - mx);
    float inv = 1.f / sum;
    for (int m2 = 0; m2 < NCTX_; ++m2) {
      float p = __expf(S[lane][m2] - mx) * inv;
      P[lane][m2] = f2bf(p);
      if (m2 >= 1)
        atomicAdd(&imp_c[(g * N_ + ng) * NCB_ + (m2 - 1)], p * (1.f / (float)REP_));
    }
  }
  __syncthreads();

  // out = P(16x32) @ Cv(32x64); Cv is dim-major so B-fragments are contiguous
  for (int nt = 0; nt < 4; ++nt) {
    v8f acc = {};
    v16bf a = load_A(&P[0][0], NCTX_);
    v16bf b = load_Bt(cvbT + (g * DH_ + nt * 16) * NCTX_, NCTX_);
    acc = wmma_bf16(a, b, acc);
    int dc = nt * 16 + (lane & 15);
    int mb = (lane >> 4) * 8;
#pragma unroll
    for (int r = 0; r < 8; ++r)
      c_out[(h * N_ + (n0 + mb + r)) * DH_ + dc] = acc[r];
  }
}

// ---------------- block importance + top-8 selection ----------------
__global__ void k_select(const float* __restrict__ imp_c, int* __restrict__ sel_idx) {
  int i = blockIdx.x * blockDim.x + threadIdx.x;
  if (i >= KVH_ * N_) return;
  int g = i / N_, n = i % N_;
  float imp[NSB_];
#pragma unroll
  for (int s = 0; s < NSB_; ++s) imp[s] = 0.f;
  const float* ic = imp_c + (g * N_ + n) * NCB_;
  for (int m = 0; m < NCB_; ++m) {
    float v = ic[m];
    int clo = m * STRIDE_, chi = clo + BLK_;
    for (int s = 0; s < NSB_; ++s) {
      int slo = s * BLK_;
      int ov = min(chi, slo + BLK_) - max(clo, slo);
      if (ov > 0) imp[s] += v * (float)ov * (1.f / (float)BLK_);
    }
  }
  int own = n / BLK_;
#pragma unroll
  for (int s = 0; s < NSB_; ++s) {
    if (s * BLK_ > n) imp[s] = NEGF;
    if (s == own) imp[s] = 1e9f;
  }
  int* out = sel_idx + (g * N_ + n) * NSEL_;
  bool used[NSB_];
#pragma unroll
  for (int s = 0; s < NSB_; ++s) used[s] = false;
  for (int j = 0; j < NSEL_; ++j) {
    int best = 0;
    float bv = -3.4e38f;
    for (int s = 0; s < NSB_; ++s)
      if (!used[s] && imp[s] > bv) { bv = imp[s]; best = s; }
    used[best] = true;
    out[j] = (bv > NEGF * 0.5f) ? best : -1;
  }
}

// ---------------- fine + sliding attention: one wave per (g, n) ----------------
__global__ void __launch_bounds__(32) k_attn_fs(const bf16* __restrict__ rq,
                                                const bf16* __restrict__ rk,
                                                const bf16* __restrict__ vT,
                                                const int* __restrict__ sel_idx,
                                                float* __restrict__ f_out,
                                                float* __restrict__ s_out) {
  int g = blockIdx.x >> 10;
  int n = blockIdx.x & (N_ - 1);
  int lane = threadIdx.x;
  __shared__ __attribute__((aligned(32))) float S[16][NSEL_ * BLK_];  // 32 KB
  __shared__ __attribute__((aligned(32))) bf16 P[16][NSEL_ * BLK_];   // 16 KB

  // A fragments: rows 0..3 = rep heads (roped q), rows 4..15 = zero pad
  int m = lane & 15;
  int kh8 = (lane >> 4) << 4;
  v16bf a0, a1;
  {
    const bf16* qrow = rq + ((g * REP_ + (m < REP_ ? m : 0)) * N_ + n) * DH_;
#pragma unroll
    for (int e = 0; e < 16; ++e) {
      int k = e + kh8;
      a0[e] = (m < REP_) ? qrow[k] : f2bf(0.f);
      a1[e] = (m < REP_) ? qrow[32 + k] : f2bf(0.f);
    }
  }

  // ---- fine scores over 8 selected blocks (keys contiguous per block) ----
  const int* sel = sel_idx + (g * N_ + n) * NSEL_;
  for (int j = 0; j < NSEL_; ++j) {
    int sb = sel[j];
    if (sb >= 0) {
      int tok0 = sb * BLK_;
      for (int nt = 0; nt < 4; ++nt) {
        v8f acc = {};
        const bf16* kbase = rk + (g * N_ + tok0 + nt * 16) * DH_;
        acc = wmma_bf16(a0, load_Bt(kbase, DH_), acc);
        acc = wmma_bf16(a1, load_Bt(kbase + 32, DH_), acc);
        int kcol = nt * 16 + (lane & 15);
        int t = tok0 + kcol;
        int mb = (lane >> 4) * 8;
#pragma unroll
        for (int r = 0; r < 8; ++r)
          S[mb + r][j * BLK_ + kcol] = (t <= n) ? acc[r] * SCALE_ : NEGF;
      }
    } else {
      for (int c = lane; c < BLK_; c += 32)
#pragma unroll
        for (int qi = 0; qi < 16; ++qi) S[qi][j * BLK_ + c] = NEGF;
    }
  }
  __syncthreads();

  // ---- fine softmax (rows 0..3, 8 lanes per row) ----
  {
    int row = lane >> 3, sub = lane & 7;
    float mx = -3.4e38f;
    for (int c = sub; c < NSEL_ * BLK_; c += 8) mx = fmaxf(mx, S[row][c]);
    for (int o = 1; o < 8; o <<= 1) mx = fmaxf(mx, __shfl_xor(mx, o, 32));
    float sum = 0.f;
    for (int c = sub; c < NSEL_ * BLK_; c += 8) sum += __expf(S[row][c] - mx);
    for (int o = 1; o < 8; o <<= 1) sum += __shfl_xor(sum, o, 32);
    float inv = 1.f / sum;
    for (int c = sub; c < NSEL_ * BLK_; c += 8)
      P[row][c] = f2bf(__expf(S[row][c] - mx) * inv);
  }
  for (int c = lane; c < NSEL_ * BLK_; c += 32)
#pragma unroll
    for (int qi = REP_; qi < 16; ++qi) P[qi][c] = f2bf(0.f);
  __syncthreads();

  // ---- fine PV: P(16x512) @ V(512x64); vT dim-major -> contiguous B runs ----
  for (int nt = 0; nt < 4; ++nt) {
    v8f acc = {};
    for (int j = 0; j < NSEL_; ++j) {
      int sb = sel[j];
      if (sb < 0) continue;
      int tok0 = sb * BLK_;
#pragma unroll
      for (int ks = 0; ks < 2; ++ks) {
        v16bf a = load_A(&P[0][0] + j * BLK_ + ks * 32, NSEL_ * BLK_);
        v16bf b = load_Bt(vT + (g * DH_ + nt * 16) * N_ + tok0 + ks * 32, N_);
        acc = wmma_bf16(a, b, acc);
      }
    }
    int dc = nt * 16 + (lane & 15);
    int mb = (lane >> 4) * 8;
#pragma unroll
    for (int r = 0; r < 8; ++r) {
      int qi = mb + r;
      if (qi < REP_) f_out[((g * REP_ + qi) * N_ + n) * DH_ + dc] = acc[r];
    }
  }
  __syncthreads();

  // ---- sliding-window scores (keys = rows n-63..n, clamped; masked below 0) ----
  for (int nt = 0; nt < 4; ++nt) {
    v8f acc = {};
#pragma unroll
    for (int ks = 0; ks < 2; ++ks) {
      int ncol = lane & 15;
      int kh16 = (lane >> 4) << 4;
      int i2 = nt * 16 + ncol;
      int w = n - (WIN_ - 1) + i2;
      int rowi = w < 0 ? 0 : w;
      v16bf b = *(const v16bf*)(rk + (g * N_ + rowi) * DH_ + ks * 32 + kh16);
      acc = wmma_bf16(ks == 0 ? a0 : a1, b, acc);
    }
    int icol = nt * 16 + (lane & 15);
    int w = n - (WIN_ - 1) + icol;
    int mb = (lane >> 4) * 8;
#pragma unroll
    for (int r = 0; r < 8; ++r)
      S[mb + r][icol] = (w >= 0) ? acc[r] * SCALE_ : NEGF;
  }
  __syncthreads();
  {
    int row = lane >> 3, sub = lane & 7;
    float mx = -3.4e38f;
    for (int c = sub; c < WIN_; c += 8) mx = fmaxf(mx, S[row][c]);
    for (int o = 1; o < 8; o <<= 1) mx = fmaxf(mx, __shfl_xor(mx, o, 32));
    float sum = 0.f;
    for (int c = sub; c < WIN_; c += 8) sum += __expf(S[row][c] - mx);
    for (int o = 1; o < 8; o <<= 1) sum += __shfl_xor(sum, o, 32);
    float inv = 1.f / sum;
    for (int c = sub; c < WIN_; c += 8)
      P[row][c] = f2bf(__expf(S[row][c] - mx) * inv);
  }
  for (int c = lane; c < WIN_; c += 32)
#pragma unroll
    for (int qi = REP_; qi < 16; ++qi) P[qi][c] = f2bf(0.f);
  __syncthreads();

  // ---- sliding PV ----
  bool aligned_win = (n >= WIN_ - 1);
  for (int nt = 0; nt < 4; ++nt) {
    v8f acc = {};
#pragma unroll
    for (int ks = 0; ks < 2; ++ks) {
      v16bf a = load_A(&P[0][0] + ks * 32, NSEL_ * BLK_);
      int ncol = lane & 15;
      int kh16 = (lane >> 4) << 4;
      v16bf b;
      if (aligned_win) {
        // contiguous token run; may be unaligned to 32B -> element loads of a
        // contiguous run (compiler coalesces); row base is per-lane dim row
        const bf16* vr = vT + (g * DH_ + nt * 16 + ncol) * N_ + (n - (WIN_ - 1)) + ks * 32 + kh16;
#pragma unroll
        for (int e = 0; e < 16; ++e) b[e] = vr[e];
      } else {
#pragma unroll
        for (int e = 0; e < 16; ++e) {
          int i2 = ks * 32 + e + kh16;
          int w = n - (WIN_ - 1) + i2;
          int rowi = w < 0 ? 0 : w;
          b[e] = vT[(g * DH_ + nt * 16 + ncol) * N_ + rowi];
        }
      }
      acc = wmma_bf16(a, b, acc);
    }
    int dc = nt * 16 + (lane & 15);
    int mb = (lane >> 4) * 8;
#pragma unroll
    for (int r = 0; r < 8; ++r) {
      int qi = mb + r;
      if (qi < REP_) s_out[((g * REP_ + qi) * N_ + n) * DH_ + dc] = acc[r];
    }
  }
}

// ---------------- gates = sigmoid(xn @ Wg + bg) ----------------
__global__ void k_gates(const float* __restrict__ xn, const float* __restrict__ Wg,
                        const float* __restrict__ bg, float* __restrict__ gates) {
  int n = blockIdx.x;
  int j = threadIdx.x;
  if (j >= 3 * H_) return;
  float acc = bg[j];
  const float* row = xn + n * DIM_;
  for (int k = 0; k < DIM_; ++k) acc += row[k] * Wg[k * (3 * H_) + j];
  gates[n * (3 * H_) + j] = 1.f / (1.f + __expf(-acc));
}

// ---------------- gate-combine into head-major activations (bf16) ----------------
__global__ void k_combine(const float* __restrict__ c_out, const float* __restrict__ f_out,
                          const float* __restrict__ s_out, const float* __restrict__ gates,
                          bf16* __restrict__ oh) {
  int h = blockIdx.x % H_;
  int n = blockIdx.x / H_;
  int d = threadIdx.x;
  float g0 = gates[n * (3 * H_) + h * 3 + 0];
  float g1 = gates[n * (3 * H_) + h * 3 + 1];
  float g2 = gates[n * (3 * H_) + h * 3 + 2];
  int off = (h * N_ + n) * DH_ + d;
  float v = g0 * c_out[off] + g1 * f_out[off] + g2 * s_out[off];
  oh[n * (H_ * DH_) + h * DH_ + d] = f2bf(v);
}

// ---------------- host launcher ----------------
extern "C" void kernel_launch(void* const* d_in, const int* in_sizes, int n_in,
                              void* d_out, int out_size, void* d_ws, size_t ws_size,
                              hipStream_t stream) {
  const float* x      = (const float*)d_in[0];
  // d_in[1] = freqs_cis (unused by reference path)
  const float* rms_g  = (const float*)d_in[2];
  const float* Wqkv   = (const float*)d_in[3];
  const float* k_pos  = (const float*)d_in[4];
  const float* v_pos  = (const float*)d_in[5];
  const float* Wck    = (const float*)d_in[6];
  const float* bck    = (const float*)d_in[7];
  const float* Wcv    = (const float*)d_in[8];
  const float* bcv    = (const float*)d_in[9];
  const float* mem_ck = (const float*)d_in[10];
  const float* mem_cv = (const float*)d_in[11];
  const float* Wg     = (const float*)d_in[12];
  const float* bg     = (const float*)d_in[13];
  const float* Wo     = (const float*)d_in[14];
  float* out = (float*)d_out;
  (void)in_sizes; (void)n_in; (void)out_size; (void)ws_size;

  char* wsp = (char*)d_ws;
  auto alloc = [&](size_t bytes) -> void* {
    void* p = (void*)wsp;
    wsp += (bytes + 255) & ~(size_t)255;
    return p;
  };
  float* xn_f32   = (float*)alloc((size_t)N_ * DIM_ * 4);
  bf16*  xn_bf    = (bf16*)alloc((size_t)N_ * DIM_ * 2);
  float* qkv_f32  = (float*)alloc((size_t)N_ * QKVW_ * 4);
  bf16*  WqkvT_bf = (bf16*)alloc((size_t)DIM_ * QKVW_ * 2);
  bf16*  WckT_bf  = (bf16*)alloc((size_t)BLK_ * DH_ * DH_ * 2);
  bf16*  WcvT_bf  = (bf16*)alloc((size_t)BLK_ * DH_ * DH_ * 2);
  bf16*  WoT_bf   = (bf16*)alloc((size_t)DIM_ * DIM_ * 2);
  bf16*  rq_bf    = (bf16*)alloc((size_t)H_ * N_ * DH_ * 2);
  bf16*  q_bf     = (bf16*)alloc((size_t)H_ * N_ * DH_ * 2);
  bf16*  rk_bf    = (bf16*)alloc((size_t)KVH_ * N_ * DH_ * 2);
  bf16*  vT_bf    = (bf16*)alloc((size_t)KVH_ * N_ * DH_ * 2);
  bf16*  ckin     = (bf16*)alloc((size_t)KVH_ * NCB_ * BLK_ * DH_ * 2);
  bf16*  cvin     = (bf16*)alloc((size_t)KVH_ * NCB_ * BLK_ * DH_ * 2);
  float* cktmp    = (float*)alloc((size_t)KVH_ * NCB_ * DH_ * 4);
  float* cvtmp    = (float*)alloc((size_t)KVH_ * NCB_ * DH_ * 4);
  bf16*  ckb      = (bf16*)alloc((size_t)KVH_ * NCTX_ * DH_ * 2);
  bf16*  cvbT     = (bf16*)alloc((size_t)KVH_ * NCTX_ * DH_ * 2);
  float* imp_c    = (float*)alloc((size_t)KVH_ * N_ * NCB_ * 4);
  int*   sel      = (int*)alloc((size_t)KVH_ * N_ * NSEL_ * 4);
  float* c_out    = (float*)alloc((size_t)H_ * N_ * DH_ * 4);
  float* f_out    = (float*)alloc((size_t)H_ * N_ * DH_ * 4);
  float* s_out    = (float*)alloc((size_t)H_ * N_ * DH_ * 4);
  float* gates    = (float*)alloc((size_t)N_ * 3 * H_ * 4);
  bf16*  oh       = (bf16*)alloc((size_t)N_ * H_ * DH_ * 2);

  // weight conversion + transpose to N-major (B^T layout for WMMA loads)
  k_cvt_t<<<(DIM_ * QKVW_ + 255) / 256, 256, 0, stream>>>(Wqkv, WqkvT_bf, DIM_, QKVW_);
  k_cvt_t<<<(BLK_ * DH_ * DH_ + 255) / 256, 256, 0, stream>>>(Wck, WckT_bf, BLK_ * DH_, DH_);
  k_cvt_t<<<(BLK_ * DH_ * DH_ + 255) / 256, 256, 0, stream>>>(Wcv, WcvT_bf, BLK_ * DH_, DH_);
  k_cvt_t<<<(DIM_ * DIM_ + 255) / 256, 256, 0, stream>>>(Wo, WoT_bf, DIM_, DIM_);
  k_zero<<<(KVH_ * N_ * NCB_ + 255) / 256, 256, 0, stream>>>(imp_c, KVH_ * N_ * NCB_);

  // RMSNorm
  k_rms<<<N_, 256, 0, stream>>>(x, rms_g, xn_f32, xn_bf);

  // qkv = xn @ Wqkv  (1024 x 768 x 512)
  {
    int waves = (N_ / 16) * (QKVW_ / 16);
    k_gemm<<<(waves + 3) / 4, 128, 0, stream>>>(xn_bf, WqkvT_bf, nullptr, qkv_f32, N_, QKVW_, DIM_);
  }

  // RoPE / head split + compressed-projection inputs
  k_rope<<<N_ * (H_ + KVH_), 32, 0, stream>>>(qkv_f32, rq_bf, q_bf, rk_bf, vT_bf);
  k_build_cin<<<KVH_ * NCB_, 256, 0, stream>>>(qkv_f32, k_pos, v_pos, ckin, cvin);

  // compressed K/V projections (62 x 64 x 4096)
  {
    int M = KVH_ * NCB_;
    int waves = ((M + 15) / 16) * (DH_ / 16);
    k_gemm<<<(waves + 3) / 4, 128, 0, stream>>>(ckin, WckT_bf, bck, cktmp, M, DH_, BLK_ * DH_);
    k_gemm<<<(waves + 3) / 4, 128, 0, stream>>>(cvin, WcvT_bf, bcv, cvtmp, M, DH_, BLK_ * DH_);
  }
  k_assemble<<<(KVH_ * NCTX_ * DH_ + 255) / 256, 256, 0, stream>>>(cktmp, cvtmp, mem_ck, mem_cv,
                                                                   ckb, cvbT);

  // compressed attention + importance
  k_cattn<<<KVH_ * REP_ * (N_ / 16), 32, 0, stream>>>(q_bf, ckb, cvbT, c_out, imp_c);

  // top-8 block selection
  k_select<<<(KVH_ * N_ + 127) / 128, 128, 0, stream>>>(imp_c, sel);

  // fine + sliding attention
  k_attn_fs<<<KVH_ * N_, 32, 0, stream>>>(rq_bf, rk_bf, vT_bf, sel, f_out, s_out);

  // gates + combine
  k_gates<<<N_, 32, 0, stream>>>(xn_f32, Wg, bg, gates);
  k_combine<<<N_ * H_, DH_, 0, stream>>>(c_out, f_out, s_out, gates, oh);

  // final projection: out = oh @ Wo (1024 x 512 x 512) -> f32 d_out
  {
    int waves = (N_ / 16) * (DIM_ / 16);
    k_gemm<<<(waves + 3) / 4, 128, 0, stream>>>(oh, WoT_bf, nullptr, out, N_, DIM_, DIM_);
  }
}
